// Qwen3MoeSparseMoeBlock_1090921693845
// MI455X (gfx1250) — compile-verified, matrix operand order checked by reading, not run on hardware
//
#include <hip/hip_runtime.h>
#include <hip/hip_bf16.h>
#include <cstdint>
#include <cstddef>

#define HIDDEN      2048
#define INTER       768
#define NUM_EXPERTS 32
#define NUM_TOKENS  2048

typedef __attribute__((ext_vector_type(16))) _Float16 v16h;
typedef __attribute__((ext_vector_type(8)))  float    v8f;
typedef __attribute__((ext_vector_type(4)))  unsigned int v4u;
typedef __attribute__((ext_vector_type(2)))  unsigned int v2u;
typedef __attribute__((ext_vector_type(4)))  float    v4f;

// A/B fragment containers (trivially-copyable ext-vector unions).
union AFrag { v16h h; v4u q[2]; };
union BFrag { v16h h; _Float16 e[16]; };

// K-index for f16 WMMA A/B fragments (16-bit 16x32 layout, CDNA5 ISA 7.12.2):
// lanes 0-15 hold K = {0..7, 16..23}; lanes 16-31 hold K = {8..15, 24..31}.
__device__ __forceinline__ int kmap(int j, int hi) {
    const int v = j >> 1, p = j & 1;
    return (v < 4) ? (hi * 8 + 2 * v + p) : (16 + hi * 8 + 2 * (v - 4) + p);
}

__device__ __forceinline__ float fast_sigmoid(float g) {
    // v_exp_f32 + v_rcp_f32 instead of the IEEE div_scale/fma fixup chain.
    return __builtin_amdgcn_rcpf(1.f + __expf(-g));
}

// ---------------------------------------------------------------- k_zero ----
__global__ __launch_bounds__(256) void k_zero(float* __restrict__ out,
                                              int* __restrict__ cnt) {
    const int i = blockIdx.x * 256 + threadIdx.x;     // 4096 * 256 threads
    const v4f z = {0.f, 0.f, 0.f, 0.f};
    ((v4f*)out)[i] = z;                               // 16 MB of zeros
    if (blockIdx.x == 0 && threadIdx.x < NUM_EXPERTS) cnt[threadIdx.x] = 0;
}

// -------------------------------------------------------------- k_convert ---
__global__ __launch_bounds__(256) void k_convert(const float* __restrict__ x,
                                                 _Float16* __restrict__ x16) {
    const int i = blockIdx.x * 256 + threadIdx.x;     // 1M threads, 4 elems each
    const v4f v = ((const v4f*)x)[i];
    union { _Float16 h[4]; v2u u; } r;
    r.h[0] = (_Float16)v.x; r.h[1] = (_Float16)v.y;
    r.h[2] = (_Float16)v.z; r.h[3] = (_Float16)v.w;
    ((v2u*)x16)[i] = r.u;
}

// --------------------------------------------------------------- k_router ---
// One wave32 per token; lane == expert. Softmax over 32 logits, top-4 with
// lowest-index tie-break, renormalize, append (token, weight) to expert lists.
__global__ __launch_bounds__(32) void k_router(const float* __restrict__ x,
                                               const float* __restrict__ wr,
                                               int* __restrict__ cnt,
                                               int* __restrict__ tok,
                                               float* __restrict__ wgt) {
    const int t = blockIdx.x;
    const int e = threadIdx.x;                 // 0..31
    const v4f* xr4 = (const v4f*)(x + (size_t)t * HIDDEN);

    float acc = 0.f;
    for (int h4 = 0; h4 < HIDDEN / 4; ++h4) {
        const v4f xv = xr4[h4];                // broadcast across the wave
        const int h = h4 * 4;
        acc = fmaf(xv.x, wr[(h + 0) * NUM_EXPERTS + e], acc);
        acc = fmaf(xv.y, wr[(h + 1) * NUM_EXPERTS + e], acc);
        acc = fmaf(xv.z, wr[(h + 2) * NUM_EXPERTS + e], acc);
        acc = fmaf(xv.w, wr[(h + 3) * NUM_EXPERTS + e], acc);
    }

    // softmax across the wave
    float m = acc;
    for (int off = 16; off; off >>= 1) m = fmaxf(m, __shfl_xor(m, off, 32));
    const float p = __expf(acc - m);
    float s = p;
    for (int off = 16; off; off >>= 1) s += __shfl_xor(s, off, 32);
    const float rw = p / s;

    // iterative top-4 (value desc, index asc on ties)
    float v = rw, sum4 = 0.f;
    bool sel = false;
    for (int it = 0; it < 4; ++it) {
        float bv = v; int bl = e;
        for (int off = 16; off; off >>= 1) {
            const float ov = __shfl_xor(bv, off, 32);
            const int   ol = __shfl_xor(bl, off, 32);
            if (ov > bv || (ov == bv && ol < bl)) { bv = ov; bl = ol; }
        }
        sum4 += bv;
        if (e == bl) { sel = true; v = -1.f; }
    }
    if (sel) {
        const int pos = atomicAdd(&cnt[e], 1);
        tok[e * NUM_TOKENS + pos] = t;
        wgt[e * NUM_TOKENS + pos] = rw / sum4;
    }
}

// ------------------------------------------------------------- k_pad_scan ---
// Pad each expert's list to a multiple of 64 (token 0, weight 0) and compute
// compact h-buffer row offsets.
__global__ void k_pad_scan(const int* __restrict__ cnt, int* __restrict__ offs,
                           int* __restrict__ tok, float* __restrict__ wgt) {
    const int e = threadIdx.x;                 // 32 threads
    const int c = cnt[e];
    const int pc = (c + 63) & ~63;
    for (int i = c; i < pc; ++i) {
        tok[e * NUM_TOKENS + i] = 0;
        wgt[e * NUM_TOKENS + i] = 0.f;
    }
    __shared__ int pads[NUM_EXPERTS];
    pads[e] = pc;
    __syncthreads();
    if (e == 0) {
        int a = 0;
        for (int i = 0; i < NUM_EXPERTS; ++i) { offs[i] = a; a += pads[i]; }
    }
}

// -------------------------------------------------------------- k_gate_up ---
// grid = (E*32 tiles, INTER/128); block = 256 (8 waves). Wave owns a 64x16
// output tile (four 16x16 WMMA row-tiles) for BOTH gate and up: each B
// fragment pair feeds 8 WMMAs -> ~5 vmem per WMMA and 4x weight reuse.
__global__ __launch_bounds__(256) void k_gate_up(
    const _Float16* __restrict__ x16,
    const float* __restrict__ w_gate, const float* __restrict__ w_up,
    const int* __restrict__ cnt, const int* __restrict__ offs,
    const int* __restrict__ tok, _Float16* __restrict__ h16) {

    const int e    = blockIdx.x >> 5;
    const int tile = blockIdx.x & 31;
    if (tile * 64 >= cnt[e]) return;           // block-uniform: EXEC stays full

    const int lane = threadIdx.x & 31;
    const int wv   = threadIdx.x >> 5;
    const int r    = lane & 15;                // A row / B column
    const int hi   = lane >> 4;
    const int n0   = blockIdx.y * 128 + wv * 16;

    const int rbase = e * NUM_TOKENS + tile * 64;
    const v4u* xq[4];
#pragma unroll
    for (int i = 0; i < 4; ++i) {
        const int token = tok[rbase + 16 * i + r];
        xq[i] = (const v4u*)(x16 + (size_t)token * HIDDEN);
    }
    const float* wg = w_gate + (size_t)e * HIDDEN * INTER + n0 + r;
    const float* wu = w_up   + (size_t)e * HIDDEN * INTER + n0 + r;

    v8f accg[4] = {}; v8f accu[4] = {};
    for (int k0 = 0; k0 < HIDDEN; k0 += 32) {
        AFrag a[4];
#pragma unroll
        for (int i = 0; i < 4; ++i) {
            a[i].q[0] = xq[i][(k0 >> 3) + hi];     // 16B: K = hi*8 .. hi*8+7
            a[i].q[1] = xq[i][(k0 >> 3) + hi + 2]; // 16B: K = 16+hi*8 ..
        }
        BFrag bg, bu;
#pragma unroll
        for (int j = 0; j < 16; ++j) {
            const int K = kmap(j, hi);
            const size_t off = (size_t)(k0 + K) * INTER;
            bg.e[j] = (_Float16)wg[off];       // coalesced over lanes 0-15
            bu.e[j] = (_Float16)wu[off];
        }
        // Speculative prefetch of the next k-step's weight rows: translation
        // faults are dropped, so no guard (keeps the hot loop branch-free).
        __builtin_prefetch(wg + (size_t)(k0 + 32 + hi * 8) * INTER, 0, 1);
        __builtin_prefetch(wu + (size_t)(k0 + 32 + hi * 8) * INTER, 0, 1);
#pragma unroll
        for (int i = 0; i < 4; ++i)
            accg[i] = __builtin_amdgcn_wmma_f32_16x16x32_f16(
                false, a[i].h, false, bg.h, (short)0, accg[i], false, false);
#pragma unroll
        for (int i = 0; i < 4; ++i)
            accu[i] = __builtin_amdgcn_wmma_f32_16x16x32_f16(
                false, a[i].h, false, bu.h, (short)0, accu[i], false, false);
    }

    // SwiGLU on f32 accumulators, store h as f16 (compact per-expert rows).
    _Float16* hout = h16 + ((size_t)offs[e] + tile * 64) * INTER + n0 + r;
#pragma unroll
    for (int i = 0; i < 4; ++i) {
#pragma unroll
        for (int v = 0; v < 8; ++v) {
            const int M = 16 * i + hi * 8 + v;
            const float g = accg[i][v], u = accu[i][v];
            hout[(size_t)M * INTER] = (_Float16)(g * fast_sigmoid(g) * u);
        }
    }
}

// ---------------------------------------------------------------- k_down ----
// grid = (E*32 tiles, HIDDEN/128); 64x16 wave tile, K over 768, then
// weighted atomic scatter-add into the output.
__global__ __launch_bounds__(256) void k_down(
    const _Float16* __restrict__ h16, const float* __restrict__ w_down,
    const int* __restrict__ cnt, const int* __restrict__ offs,
    const int* __restrict__ tok, const float* __restrict__ wgt,
    float* __restrict__ out) {

    const int e    = blockIdx.x >> 5;
    const int tile = blockIdx.x & 31;
    if (tile * 64 >= cnt[e]) return;

    const int lane = threadIdx.x & 31;
    const int wv   = threadIdx.x >> 5;
    const int r    = lane & 15;
    const int hi   = lane >> 4;
    const int n0   = blockIdx.y * 128 + wv * 16;

    const v4u* hq = (const v4u*)(h16 + ((size_t)offs[e] + tile * 64) * INTER);
    const float* wd = w_down + (size_t)e * INTER * HIDDEN + n0 + r;

    v8f acc[4] = {};
    for (int k0 = 0; k0 < INTER; k0 += 32) {
        AFrag a[4];
#pragma unroll
        for (int i = 0; i < 4; ++i) {
            const int rowq = (16 * i + r) * (INTER / 8);   // 16B units
            a[i].q[0] = hq[rowq + (k0 >> 3) + hi];
            a[i].q[1] = hq[rowq + (k0 >> 3) + hi + 2];
        }
        BFrag b;
#pragma unroll
        for (int j = 0; j < 16; ++j) {
            const int K = kmap(j, hi);
            b.e[j] = (_Float16)wd[(size_t)(k0 + K) * HIDDEN];
        }
        __builtin_prefetch(wd + (size_t)(k0 + 32 + hi * 8) * HIDDEN, 0, 1);
#pragma unroll
        for (int i = 0; i < 4; ++i)
            acc[i] = __builtin_amdgcn_wmma_f32_16x16x32_f16(
                false, a[i].h, false, b.h, (short)0, acc[i], false, false);
    }

    const int base = e * NUM_TOKENS + tile * 64;
#pragma unroll
    for (int i = 0; i < 4; ++i) {
#pragma unroll
        for (int v = 0; v < 8; ++v) {
            const int M = 16 * i + hi * 8 + v;
            const int   t = tok[base + M];
            const float w = wgt[base + M];
            atomicAdd(out + (size_t)t * HIDDEN + n0 + r, acc[i][v] * w);
        }
    }
}

// ------------------------------------------------------------------ launch --
extern "C" void kernel_launch(void* const* d_in, const int* in_sizes, int n_in,
                              void* d_out, int out_size, void* d_ws, size_t ws_size,
                              hipStream_t stream) {
    const float* x        = (const float*)d_in[0];
    const float* w_router = (const float*)d_in[1];
    const float* w_gate   = (const float*)d_in[2];
    const float* w_up     = (const float*)d_in[3];
    const float* w_down   = (const float*)d_in[4];
    float* out = (float*)d_out;

    // workspace layout (bytes)
    char* ws = (char*)d_ws;
    int*      cnt  = (int*)(ws + 0);                         // 32 ints
    int*      offs = (int*)(ws + 128);                       // 32 ints
    int*      tok  = (int*)(ws + 512);                       // 32*2048 ints
    float*    wgt  = (float*)(ws + 512 + (size_t)NUM_EXPERTS * NUM_TOKENS * 4);
    _Float16* x16  = (_Float16*)(ws + 524800);               // 2048*2048 f16
    _Float16* h16  = (_Float16*)(ws + 8913408);              // <=10208*768 f16 (~24.6MB total)

    k_zero    <<<4096, 256, 0, stream>>>(out, cnt);
    k_router  <<<NUM_TOKENS, 32, 0, stream>>>(x, w_router, cnt, tok, wgt);
    k_convert <<<4096, 256, 0, stream>>>(x, x16);
    k_pad_scan<<<1, 32, 0, stream>>>(cnt, offs, tok, wgt);
    k_gate_up <<<dim3(NUM_EXPERTS * 32, INTER / 128), 256, 0, stream>>>(
        x16, w_gate, w_up, cnt, offs, tok, h16);
    k_down    <<<dim3(NUM_EXPERTS * 32, HIDDEN / 128), 256, 0, stream>>>(
        h16, w_down, cnt, offs, tok, wgt, out);
}
